// ForcingFunction_74543452389768
// MI455X (gfx1250) — compile-verified, matrix operand order checked by reading, not run on hardware
//
#include <hip/hip_runtime.h>
#include <hip/hip_bf16.h>
#include <stdint.h>

typedef __attribute__((ext_vector_type(16))) _Float16 v16h;
typedef __attribute__((ext_vector_type(8)))  float    v8f;

#define BB 8
#define HH 512
#define WW 512
#define NPIX (BB * HH * WW)   // 2,097,152
#define STRIPW 128            // pixels per block strip (8 waves x 16-pixel tile)
#define T0W 132               // staged strip width incl. 5x5 halo (128 + 4)
#define TEW 132               // staged strip width for 3x3 halo (128 + 2, padded)

// k-index mapping for the 16-bit A/B operand layout (16x32 / 32x16, wave32):
// lanes 0-15 hold K={0..7,16..23} in halves 0..15; lanes 16-31 hold K={8..15,24..31}.
__device__ __forceinline__ int kmap(int lane, int h) {
    int base = (h < 8) ? 0 : 16;
    int grp  = (lane >= 16) ? 8 : 0;
    return base + grp + (h & 7);
}

// ---------------------------------------------------------------------------
// K1: edges = softsign( sum_c relu( pointwise( depthwise3x3(image) ) ) )
// ---------------------------------------------------------------------------
__global__ void edges_kernel(const float* __restrict__ image,
                             const float* __restrict__ Kdw,   // (3,3,1,8)
                             const float* __restrict__ Kpw,   // (1,1,8,8)
                             float* __restrict__ edges) {
    long idx = (long)blockIdx.x * blockDim.x + threadIdx.x;
    if (idx >= NPIX) return;
    int b = (int)(idx / (HH * WW));
    int r = (int)(idx % (HH * WW));
    int y = r / WW, x = r % WW;

    float dw[8] = {0.f,0.f,0.f,0.f,0.f,0.f,0.f,0.f};
    for (int dy = -1; dy <= 1; ++dy)
        for (int dx = -1; dx <= 1; ++dx) {
            int yy = y + dy, xx = x + dx;
            if (yy < 0 || yy >= HH || xx < 0 || xx >= WW) continue;
            float v = image[((long)b * HH + yy) * WW + xx];
            int tap = (dy + 1) * 3 + (dx + 1);
            #pragma unroll
            for (int c = 0; c < 8; ++c) dw[c] += v * Kdw[tap * 8 + c];
        }
    float s = 0.f;
    #pragma unroll
    for (int n = 0; n < 8; ++n) {
        float acc = 0.f;
        #pragma unroll
        for (int c = 0; c < 8; ++c) acc += dw[c] * Kpw[c * 8 + n];
        s += fmaxf(acc, 0.f);
    }
    edges[idx] = s / (1.f + fabsf(s));
}

// ---------------------------------------------------------------------------
// K2 (WMMA): grad_edges_{x,y} = sum_c relu(conv3x3(edges, K{x,y}))  (32 ch)
// Block = 128-pixel row strip (8 waves); halo tile staged to LDS once.
// ---------------------------------------------------------------------------
__global__ void grad_wmma_kernel(const float* __restrict__ edges,
                                 const float* __restrict__ Kx,   // (3,3,1,32)
                                 const float* __restrict__ Ky,   // (3,3,1,32)
                                 float* __restrict__ gx,
                                 float* __restrict__ gy) {
    __shared__ _Float16 wx[9 * 32];
    __shared__ _Float16 wy[9 * 32];
    __shared__ _Float16 tile[3 * TEW];   // rows y-1..y+1, cols xs-1..xs+128(+pad)

    for (int i = threadIdx.x; i < 288; i += blockDim.x) {
        wx[i] = (_Float16)Kx[i];
        wy[i] = (_Float16)Ky[i];
    }

    // strip coordinates: 4 strips per row, HH rows, BB batches
    const int strip = blockIdx.x;
    const int xs = (strip & 3) * STRIPW;
    const int t2 = strip >> 2;
    const int y  = t2 & (HH - 1);
    const int b  = t2 >> 9;

    // coalesced staging of edges strip (zero-padded halo)
    for (int e = threadIdx.x; e < 3 * TEW; e += blockDim.x) {
        int row = e / TEW, xi = e % TEW;
        int yy = y + row - 1, xx = xs + xi - 1;
        float v = 0.f;
        if (yy >= 0 && yy < HH && xx >= 0 && xx < WW)
            v = edges[((long)b * HH + yy) * WW + xx];
        tile[e] = (_Float16)v;
    }
    __syncthreads();

    const int lane = threadIdx.x & 31;
    const int wave = threadIdx.x >> 5;
    const int x0   = xs + wave * 16;
    const int m    = lane & 15;
    const int n    = lane & 15;

    // B operands: K=9 taps padded to 32; N=32 split into two halves
    v16h b0, b1, b2, b3;
    #pragma unroll
    for (int h = 0; h < 16; ++h) {
        int k  = kmap(lane, h);
        int kc = (k < 9) ? k : 8;          // clamp, select below (branchless)
        bool ok = (k < 9);
        _Float16 z = (_Float16)0.f;
        b0[h] = ok ? wx[kc * 32 + n]      : z;
        b1[h] = ok ? wx[kc * 32 + n + 16] : z;
        b2[h] = ok ? wy[kc * 32 + n]      : z;
        b3[h] = ok ? wy[kc * 32 + n + 16] : z;
    }

    // A operand: 16 pixels x 9-tap patches from LDS (branchless, halo staged)
    v16h a;
    #pragma unroll
    for (int h = 0; h < 16; ++h) {
        int k  = kmap(lane, h);
        int kc = (k < 9) ? k : 8;
        int dy = kc / 3, dx = kc % 3;      // 0..2 each
        _Float16 v = tile[dy * TEW + wave * 16 + m + dx];
        a[h] = (k < 9) ? v : (_Float16)0.f;
    }

    v8f c0 = {}, c1 = {}, c2 = {}, c3 = {};
    c0 = __builtin_amdgcn_wmma_f32_16x16x32_f16(false, a, false, b0, (short)0, c0, false, false);
    c1 = __builtin_amdgcn_wmma_f32_16x16x32_f16(false, a, false, b1, (short)0, c1, false, false);
    c2 = __builtin_amdgcn_wmma_f32_16x16x32_f16(false, a, false, b2, (short)0, c2, false, false);
    c3 = __builtin_amdgcn_wmma_f32_16x16x32_f16(false, a, false, b3, (short)0, c3, false, false);

    float sx[8], sy[8];
    #pragma unroll
    for (int vv = 0; vv < 8; ++vv) {
        sx[vv] = fmaxf(c0[vv], 0.f) + fmaxf(c1[vv], 0.f);
        sy[vv] = fmaxf(c2[vv], 0.f) + fmaxf(c3[vv], 0.f);
    }
    // reduce over N (lanes 0..15 and 16..31 independently)
    #pragma unroll
    for (int msk = 1; msk <= 8; msk <<= 1)
        #pragma unroll
        for (int vv = 0; vv < 8; ++vv) {
            sx[vv] += __shfl_xor(sx[vv], msk, 32);
            sy[vv] += __shfl_xor(sy[vv], msk, 32);
        }
    if ((lane & 15) == 0) {
        int mb = (lane >> 4) * 8;   // lane 0 -> rows 0..7, lane 16 -> rows 8..15
        long base = ((long)b * HH + y) * WW + x0 + mb;
        #pragma unroll
        for (int vv = 0; vv < 8; ++vv) { gx[base + vv] = sx[vv]; gy[base + vv] = sy[vv]; }
    }
}

// ---------------------------------------------------------------------------
// K3: t0 = conv5x5(u, curve_0)  1 -> 8 ch, stored f16 channel-innermost
// ---------------------------------------------------------------------------
__global__ void curve0_kernel(const float* __restrict__ u,
                              const float* __restrict__ K0,   // (5,5,1,8)
                              _Float16* __restrict__ t0) {
    long idx = (long)blockIdx.x * blockDim.x + threadIdx.x;
    if (idx >= NPIX) return;
    int b = (int)(idx / (HH * WW));
    int r = (int)(idx % (HH * WW));
    int y = r / WW, x = r % WW;

    float acc[8] = {0.f,0.f,0.f,0.f,0.f,0.f,0.f,0.f};
    for (int dy = -2; dy <= 2; ++dy)
        for (int dx = -2; dx <= 2; ++dx) {
            int yy = y + dy, xx = x + dx;
            if (yy < 0 || yy >= HH || xx < 0 || xx >= WW) continue;
            float v = u[((long)b * HH + yy) * WW + xx];
            int tap = (dy + 2) * 5 + (dx + 2);
            #pragma unroll
            for (int c = 0; c < 8; ++c) acc[c] += v * K0[tap * 8 + c];
        }
    #pragma unroll
    for (int c = 0; c < 8; ++c) t0[idx * 8 + c] = (_Float16)acc[c];
}

// ---------------------------------------------------------------------------
// K4 (WMMA): t1 = conv5x5(t0, curve_1)  8 -> 8 ch. K = 25 taps * 8 ch = 200,
// grouped as 7 chunks of (4 taps x 8 ch) = K=32 -> 7 accumulating WMMAs.
// Input tile staged into LDS via CDNA5 async global->LDS (ASYNCcnt) path.
// ---------------------------------------------------------------------------
__global__ void curve1_wmma_kernel(const _Float16* __restrict__ t0,
                                   const float* __restrict__ K1,   // (5,5,8,8)
                                   _Float16* __restrict__ t1) {
    __shared__ _Float16 w1[25 * 64];
    __shared__ _Float16 tile[5 * T0W * 8];   // rows y-2..y+2, cols xs-2.., 8 ch
    uint32_t* tile32 = (uint32_t*)tile;      // 5*T0W*4 dwords (2 ch per dword)

    for (int i = threadIdx.x; i < 1600; i += blockDim.x) w1[i] = (_Float16)K1[i];

    const int strip = blockIdx.x;
    const int xs = (strip & 3) * STRIPW;
    const int t2 = strip >> 2;
    const int y  = t2 & (HH - 1);
    const int b  = t2 >> 9;

    // zero-fill tile (halo default), then async-load in-range dwords to LDS
    const int NDW = 5 * T0W * 4;   // 2640 dwords
    for (int e = threadIdx.x; e < NDW; e += blockDim.x) tile32[e] = 0u;
    __syncthreads();

    for (int e = threadIdx.x; e < NDW; e += blockDim.x) {
        int row = e / (T0W * 4);
        int rem = e % (T0W * 4);
        int xi  = rem >> 2;        // staged column
        int cp  = rem & 3;         // channel pair
        int yy = y + row - 2, xx = xs + xi - 2;
        if (yy >= 0 && yy < HH && xx >= 0 && xx < WW) {
            const void* g = (const void*)&t0[(((long)b * HH + yy) * WW + xx) * 8 + 2 * cp];
            uint32_t la = (uint32_t)(uintptr_t)&tile32[e];
            asm volatile("global_load_async_to_lds_b32 %0, %1, off"
                         :: "v"(la), "v"(g) : "memory");
        }
    }
    asm volatile("s_wait_asynccnt 0x0" ::: "memory");
    __syncthreads();

    const int lane = threadIdx.x & 31;
    const int wave = threadIdx.x >> 5;
    const int x0   = xs + wave * 16;
    const int m    = lane & 15;
    const int n    = lane & 15;

    // B operands: 7 K-chunks of (4 taps x 8 ch); N=8 real of 16
    v16h bw[7];
    #pragma unroll
    for (int g = 0; g < 7; ++g)
        #pragma unroll
        for (int h = 0; h < 16; ++h) {
            int k   = kmap(lane, h);
            int tap = g * 4 + (k >> 3);
            int c   = k & 7;
            int tc  = (tap < 25) ? tap : 24;
            _Float16 v = w1[tc * 64 + c * 8 + n];
            bw[g][h] = (tap < 25 && n < 8) ? v : (_Float16)0.f;
        }

    v8f acc = {};
    #pragma unroll
    for (int g = 0; g < 7; ++g) {
        v16h a;
        #pragma unroll
        for (int h = 0; h < 16; ++h) {
            int k   = kmap(lane, h);
            int tap = g * 4 + (k >> 3);
            int c   = k & 7;
            int tc  = (tap < 25) ? tap : 24;
            int dy  = tc / 5, dx = tc % 5;     // 0..4
            _Float16 v = tile[(dy * T0W + wave * 16 + m + dx) * 8 + c];
            a[h] = (tap < 25) ? v : (_Float16)0.f;
        }
        acc = __builtin_amdgcn_wmma_f32_16x16x32_f16(false, a, false, bw[g], (short)0, acc, false, false);
    }

    if (n < 8) {
        int mb = (lane >= 16) ? 8 : 0;
        #pragma unroll
        for (int vv = 0; vv < 8; ++vv)
            t1[(((long)b * HH + y) * WW + x0 + mb + vv) * 8 + n] = (_Float16)acc[vv];
    }
}

// ---------------------------------------------------------------------------
// K5: ksum = (conv3x3(t1, sum_n curve_2)) * kappa_kernel
// (no relu before the channel sum -> fold 16 output channels into weights)
// ---------------------------------------------------------------------------
__global__ void kappa_pre_kernel(const _Float16* __restrict__ t1,
                                 const float* __restrict__ K2,   // (3,3,8,16)
                                 const float* __restrict__ kap,  // (512,512,1)
                                 float* __restrict__ ksum) {
    __shared__ float c2s[72];
    for (int i = threadIdx.x; i < 72; i += blockDim.x) {
        float s = 0.f;
        #pragma unroll
        for (int nn = 0; nn < 16; ++nn) s += K2[i * 16 + nn];
        c2s[i] = s;
    }
    __syncthreads();

    long idx = (long)blockIdx.x * blockDim.x + threadIdx.x;
    if (idx >= NPIX) return;
    int b = (int)(idx / (HH * WW));
    int r = (int)(idx % (HH * WW));
    int y = r / WW, x = r % WW;

    float acc = 0.f;
    for (int dy = -1; dy <= 1; ++dy)
        for (int dx = -1; dx <= 1; ++dx) {
            int yy = y + dy, xx = x + dx;
            if (yy < 0 || yy >= HH || xx < 0 || xx >= WW) continue;
            int tap = (dy + 1) * 3 + (dx + 1);
            const _Float16* p = &t1[(((long)b * HH + yy) * WW + xx) * 8];
            #pragma unroll
            for (int c = 0; c < 8; ++c) acc += (float)p[c] * c2s[tap * 8 + c];
        }
    ksum[idx] = acc * kap[(long)y * WW + x];
}

// ---------------------------------------------------------------------------
// K6: out = u + fxp*xp + fxn*xn + fyp*yp + fyn*yn + edges + conv5x5(ksum, sm)
// ---------------------------------------------------------------------------
__global__ void final_kernel(const float* __restrict__ u,
                             const float* __restrict__ edges,
                             const float* __restrict__ gx,
                             const float* __restrict__ gy,
                             const float* __restrict__ ksum,
                             const float* __restrict__ Ksm,   // (5,5,1,1)
                             float* __restrict__ out) {
    long idx = (long)blockIdx.x * blockDim.x + threadIdx.x;
    if (idx >= NPIX) return;
    int b = (int)(idx / (HH * WW));
    int r = (int)(idx % (HH * WW));
    int y = r / WW, x = r % WW;
    long base = (long)b * HH * WW;

    if (y + 1 < HH) __builtin_prefetch(u + base + (long)(y + 1) * WW + x, 0, 1);

    float uc = u[idx];
    float ul = (x > 0)       ? u[idx - 1]  : 0.f;
    float ur = (x + 1 < WW)  ? u[idx + 1]  : 0.f;
    float uu = (y > 0)       ? u[idx - WW] : 0.f;
    float ud = (y + 1 < HH)  ? u[idx + WW] : 0.f;

    float xp = uc - ul;   // conv(u, KXP)
    float xn = ur - uc;   // conv(u, KXN)
    float yn = uu - uc;   // conv(u, KYN)
    float yp = uc - ud;   // conv(u, KYP)

    float gxx = gx[idx], gyy = gy[idx];
    float fxp = fmaxf(gxx, 0.f), fxn = fminf(gxx, 0.f);
    float fyp = fmaxf(gyy, 0.f), fyn = fminf(gyy, 0.f);

    float sm = 0.f;
    for (int dy = -2; dy <= 2; ++dy)
        for (int dx = -2; dx <= 2; ++dx) {
            int yy = y + dy, xx = x + dx;
            if (yy < 0 || yy >= HH || xx < 0 || xx >= WW) continue;
            sm += ksum[base + (long)yy * WW + xx] * Ksm[(dy + 2) * 5 + (dx + 2)];
        }

    out[idx] = uc + fxp * xp + fxn * xn + fyp * yp + fyn * yn + edges[idx] + sm;
}

// ---------------------------------------------------------------------------
extern "C" void kernel_launch(void* const* d_in, const int* in_sizes, int n_in,
                              void* d_out, int out_size, void* d_ws, size_t ws_size,
                              hipStream_t stream) {
    (void)in_sizes; (void)n_in; (void)out_size; (void)ws_size;
    const float* u     = (const float*)d_in[0];
    const float* image = (const float*)d_in[1];
    const float* Kx    = (const float*)d_in[2];
    const float* Ky    = (const float*)d_in[3];
    const float* Kdw   = (const float*)d_in[4];
    const float* Kpw   = (const float*)d_in[5];
    const float* K0    = (const float*)d_in[6];
    const float* K1    = (const float*)d_in[7];
    const float* K2    = (const float*)d_in[8];
    const float* Ksm   = (const float*)d_in[9];
    const float* kap   = (const float*)d_in[10];
    float* out = (float*)d_out;

    // workspace layout
    float*     edges = (float*)d_ws;                 // NPIX f32
    float*     gx    = edges + NPIX;                 // NPIX f32
    float*     gy    = gx + NPIX;                    // NPIX f32
    float*     ksum  = gy + NPIX;                    // NPIX f32
    _Float16*  t0    = (_Float16*)(ksum + NPIX);     // NPIX*8 f16
    _Float16*  t1    = t0 + (size_t)NPIX * 8;        // NPIX*8 f16

    const int threads = 256;
    const int pixBlocks   = NPIX / threads;          // 8192
    const int stripBlocks = NPIX / STRIPW;           // 16384 (8 waves/strip)

    edges_kernel      <<<pixBlocks,   threads, 0, stream>>>(image, Kdw, Kpw, edges);
    grad_wmma_kernel  <<<stripBlocks, threads, 0, stream>>>(edges, Kx, Ky, gx, gy);
    curve0_kernel     <<<pixBlocks,   threads, 0, stream>>>(u, K0, t0);
    curve1_wmma_kernel<<<stripBlocks, threads, 0, stream>>>(t0, K1, t1);
    kappa_pre_kernel  <<<pixBlocks,   threads, 0, stream>>>(t1, K2, kap, ksum);
    final_kernel      <<<pixBlocks,   threads, 0, stream>>>(u, edges, gx, gy, ksum, Ksm, out);
}
